// G_PAM_Module_12455405158998
// MI455X (gfx1250) — matrix operand order repaired and checked by reading, not compiled, not run on hardware
//
#include <hip/hip_runtime.h>

// ---------------------------------------------------------------------------
// G-PAM chained attention for MI455X (gfx1250, wave32, WMMA bf16)
// B=4, C=64, CG=32, N=H*W=4096
// All GEMMs are NT (both operands k-contiguous). The dominant N^3 GEMM
// (~550 GFLOP) uses a 128x128 workgroup tile with double-buffered
// GLOBAL_LOAD_ASYNC_TO_LDS_B128 panel staging (ASYNCcnt) and WMMA fragments
// read from LDS (ds_load_b128), 8 WMMAs per wave per 32-k step.
// ---------------------------------------------------------------------------

#define DEV static __device__ __forceinline__

typedef __attribute__((ext_vector_type(16))) __bf16 v16bf;
typedef __attribute__((ext_vector_type(8)))  float  v8f;

constexpr int NB  = 4;     // batch
constexpr int CC  = 64;    // channels
constexpr int CGC = 32;    // guide channels
constexpr int NN  = 4096;  // H*W

constexpr int KCH    = 64;  // k-chunk staged per buffer
constexpr int APITCH = 72;  // LDS row pitch in elements (64 + 8 pad = 144B)

// float -> bf16 (round-to-nearest-even)
DEV unsigned short f2bf(float f) {
  unsigned int u = __builtin_bit_cast(unsigned int, f);
  u += 0x7FFFu + ((u >> 16) & 1u);
  return (unsigned short)(u >> 16);
}

// ---------------------------------------------------------------------------
// WMMA fragment loaders (CDNA5 ISA 7.12.2, wave32), k-contiguous layouts.
// A-operand 16x32: lane = row M=L%16;
//                  elems 0..15 = k+16*hi+0..15 (32B contiguous).
// B-operand 32x16 from k-major matrix: lane = col N=L%16;
//                  elems 0..15 = k+16*hi+0..15 (32B contiguous).
// Work for global or LDS row pointers (16B-aligned).
// ---------------------------------------------------------------------------
DEV v16bf load_a_frag(const unsigned short* rowptr, int kk, int hi) {
  union { uint4 q[2]; v16bf v; } u;
  const uint4* p = (const uint4*)(rowptr + kk + hi * 16);
  u.q[0] = p[0];
  u.q[1] = p[1];
  return u.v;
}
DEV v16bf load_bt_frag(const unsigned short* rowptr, int kk, int hi) {
  union { uint4 q[2]; v16bf v; } u;
  const uint4* p = (const uint4*)(rowptr + kk + hi * 16);
  u.q[0] = p[0];
  u.q[1] = p[1];
  return u.v;
}

DEV v8f wmma_bf16(v16bf a, v16bf b, v8f c) {
  return __builtin_amdgcn_wmma_f32_16x16x32_bf16(
      false, a, false, b, (short)0, c, false, false);
}

// ---------------------------------------------------------------------------
// CDNA5 async global->LDS copy (16B per lane), tracked by ASYNCcnt.
// (ISA 15.18.3 opcode 98 / async-tensor doc §4; no probe-confirmed builtin,
// so inline asm per bridge-doc guidance.)
// ---------------------------------------------------------------------------
DEV void async_copy_b128(unsigned lds_byte_off, const void* gptr) {
  asm volatile("global_load_async_to_lds_b128 %0, %1, off"
               :
               : "v"(lds_byte_off), "v"(gptr)
               : "memory");
}
DEV void wait_async0() {
  asm volatile("s_wait_asynccnt 0x0" ::: "memory");
}

// ---------------------------------------------------------------------------
// Kernel 1: 1x1-conv projections (bf16 outputs, consumer-friendly layouts):
//   qt,kt : (B, N, C)   qgt,kgt : (B, N, CG)   vcn : (B, C, N)
// ---------------------------------------------------------------------------
__global__ __launch_bounds__(256) void gpam_proj(
    const float* __restrict__ x, const float* __restrict__ g,
    const float* __restrict__ Wq, const float* __restrict__ bq,
    const float* __restrict__ Wk, const float* __restrict__ bk,
    const float* __restrict__ Wv, const float* __restrict__ bv,
    const float* __restrict__ Wqg, const float* __restrict__ bqg,
    const float* __restrict__ Wkg, const float* __restrict__ bkg,
    unsigned short* __restrict__ qt, unsigned short* __restrict__ kt,
    unsigned short* __restrict__ vcn, unsigned short* __restrict__ qgt,
    unsigned short* __restrict__ kgt) {
  const int idx = blockIdx.x * blockDim.x + threadIdx.x;  // b*N + n
  const int b = idx / NN;
  const int n = idx - b * NN;

  float xc[CC];
  const float* xb = x + (size_t)b * CC * NN;
#pragma unroll
  for (int c = 0; c < CC; ++c) xc[c] = xb[(size_t)c * NN + n];

  float gc[CGC];
  const float* gb = g + (size_t)b * CGC * NN;
#pragma unroll
  for (int c = 0; c < CGC; ++c) gc[c] = gb[(size_t)c * NN + n];

  for (int o = 0; o < CC; ++o) {
    float aq = bq[o], ak = bk[o], av = bv[o];
    const float* wq = Wq + (size_t)o * CC;
    const float* wk = Wk + (size_t)o * CC;
    const float* wv = Wv + (size_t)o * CC;
#pragma unroll
    for (int c = 0; c < CC; ++c) {
      aq = fmaf(wq[c], xc[c], aq);
      ak = fmaf(wk[c], xc[c], ak);
      av = fmaf(wv[c], xc[c], av);
    }
    qt[((size_t)b * NN + n) * CC + o]  = f2bf(aq);
    kt[((size_t)b * NN + n) * CC + o]  = f2bf(ak);
    vcn[((size_t)b * CC + o) * NN + n] = f2bf(av);
  }
  for (int o = 0; o < CGC; ++o) {
    float aq = bqg[o], ak = bkg[o];
    const float* wq = Wqg + (size_t)o * CGC;
    const float* wk = Wkg + (size_t)o * CGC;
#pragma unroll
    for (int c = 0; c < CGC; ++c) {
      aq = fmaf(wq[c], gc[c], aq);
      ak = fmaf(wk[c], gc[c], ak);
    }
    qgt[((size_t)b * NN + n) * CGC + o] = f2bf(aq);
    kgt[((size_t)b * NN + n) * CGC + o] = f2bf(ak);
  }
}

// ---------------------------------------------------------------------------
// Kernel 2: fused NT GEMM + row softmax (for small K: 64 / 32).
// grid = (N/16 strips, B); block = 256 (8 waves); 16 x 4096 f32 strip in LDS.
// trans=1 stores the softmax output transposed (k-major for the next GEMM).
// ---------------------------------------------------------------------------
__global__ __launch_bounds__(256) void gpam_gemm_softmax(
    const unsigned short* __restrict__ A, const unsigned short* __restrict__ BT,
    unsigned short* __restrict__ out, int K, int trans) {
  extern __shared__ __align__(16) unsigned char smem_raw[];
  float* strip = (float*)smem_raw;        // [16][4096]
  float* red   = strip + 16 * NN;         // [16][16]
  float* rowv  = red + 16 * 16;           // [16]

  const int b  = blockIdx.y;
  const int r0 = blockIdx.x * 16;

  const int tid  = threadIdx.x;
  const int lane = tid & 31;
  const int wave = tid >> 5;
  const int mrow = lane & 15;
  const int hi   = lane >> 4;

  const unsigned short* pa =
      A + (size_t)b * NN * K + (size_t)(r0 + mrow) * K;
  const unsigned short* BTb = BT + (size_t)b * NN * K;

  for (int jp = wave; jp < NN / 32; jp += 8) {
    const int c0 = jp * 32;
    const unsigned short* pb0 = BTb + (size_t)(c0 + mrow) * K;
    const unsigned short* pb1 = pb0 + (size_t)16 * K;

    v8f acc0, acc1;
#pragma unroll
    for (int r = 0; r < 8; ++r) { acc0[r] = 0.0f; acc1[r] = 0.0f; }

    for (int kk = 0; kk < K; kk += 32) {
      v16bf af = load_a_frag(pa, kk, hi);
      v16bf b0 = load_bt_frag(pb0, kk, hi);
      v16bf b1 = load_bt_frag(pb1, kk, hi);
      acc0 = wmma_bf16(af, b0, acc0);
      acc1 = wmma_bf16(af, b1, acc1);
    }
#pragma unroll
    for (int r = 0; r < 8; ++r) {
      strip[(r + hi * 8) * NN + c0 + mrow]      = acc0[r];
      strip[(r + hi * 8) * NN + c0 + 16 + mrow] = acc1[r];
    }
  }
  __syncthreads();

  const int row = tid >> 4;
  const int l16 = tid & 15;
  float* srow = strip + row * NN;

  float m = -3.402823466e38f;
  for (int i = l16; i < NN; i += 16) m = fmaxf(m, srow[i]);
  red[row * 16 + l16] = m;
  __syncthreads();
  if (l16 == 0) {
    float mm = red[row * 16];
#pragma unroll
    for (int i = 1; i < 16; ++i) mm = fmaxf(mm, red[row * 16 + i]);
    rowv[row] = mm;
  }
  __syncthreads();
  const float rm = rowv[row];

  float s = 0.0f;
  for (int i = l16; i < NN; i += 16) {
    float e = __expf(srow[i] - rm);
    srow[i] = e;
    s += e;
  }
  __syncthreads();
  red[row * 16 + l16] = s;
  __syncthreads();
  if (l16 == 0) {
    float ss = 0.0f;
#pragma unroll
    for (int i = 0; i < 16; ++i) ss += red[row * 16 + i];
    rowv[row] = ss;
  }
  __syncthreads();
  const float inv = 1.0f / rowv[row];

  unsigned short* ob = out + (size_t)b * NN * NN;
  if (trans) {
    for (int i = l16; i < NN; i += 16)
      ob[(size_t)i * NN + (r0 + row)] = f2bf(srow[i] * inv);
  } else {
    unsigned short* orow = ob + (size_t)(r0 + row) * NN;
    for (int i = l16; i < NN; i += 16) orow[i] = f2bf(srow[i] * inv);
  }
}

// ---------------------------------------------------------------------------
// Kernel 3: blocked NT GEMM for the dominant N^3 stage, async-LDS staged.
// GE = A (N x K) * BT^T, f32 output. WG tile 128x128, 8 waves (4 row-groups
// x 2 col-groups); wave = 32x64 block = 2 A-frags x 4 B-frags -> 8 WMMAs per
// k-step. A/B 128x64 panels double-buffered in LDS (4 x 18KB), staged with
// GLOBAL_LOAD_ASYNC_TO_LDS_B128 overlapped with compute on the other buffer.
// ---------------------------------------------------------------------------
__global__ __launch_bounds__(256) void gpam_gemm_blocked(
    const unsigned short* __restrict__ A, const unsigned short* __restrict__ BT,
    float* __restrict__ GE, int K) {
  __shared__ __align__(16) unsigned short smA[2][128 * APITCH];
  __shared__ __align__(16) unsigned short smB[2][128 * APITCH];

  const int b  = blockIdx.z;
  const int R0 = blockIdx.y * 128;
  const int C0 = blockIdx.x * 128;

  const int tid  = threadIdx.x;
  const int lane = tid & 31;
  const int wave = tid >> 5;
  const int mrow = lane & 15;
  const int hi   = lane >> 4;
  const int wr   = (wave >> 1) * 32;  // wave row offset in WG tile
  const int wc   = (wave & 1) * 64;   // wave col offset in WG tile

  const unsigned short* Ab = A  + (size_t)b * NN * K;
  const unsigned short* Bb = BT + (size_t)b * NN * K;

  const unsigned ldsA0 = (unsigned)(unsigned long long)(void*)&smA[0][0];
  const unsigned ldsB0 = (unsigned)(unsigned long long)(void*)&smB[0][0];
  constexpr unsigned BUFB = 128u * APITCH * 2u;  // bytes per buffer

  // Stage one 128x64 A panel + 128x64 B panel: 1024 16B segments each half,
  // 256 threads x 4 iterations; thread -> (row = li>>3, 16B seg = li&7).
  auto stage = [&](int kc, int buf) {
    const unsigned baseA = ldsA0 + (unsigned)buf * BUFB;
    const unsigned baseB = ldsB0 + (unsigned)buf * BUFB;
#pragma unroll
    for (int q = 0; q < 4; ++q) {
      const int li = q * 256 + tid;
      const int r  = li >> 3;
      const int sg = li & 7;
      const unsigned loff = (unsigned)(r * APITCH + sg * 8) * 2u;
      async_copy_b128(baseA + loff, Ab + (size_t)(R0 + r) * K + kc + sg * 8);
      async_copy_b128(baseB + loff, Bb + (size_t)(C0 + r) * K + kc + sg * 8);
    }
  };

  v8f acc[2][4];
#pragma unroll
  for (int i = 0; i < 2; ++i)
#pragma unroll
    for (int j = 0; j < 4; ++j)
#pragma unroll
      for (int r = 0; r < 8; ++r) acc[i][j][r] = 0.0f;

  stage(0, 0);
  wait_async0();
  __syncthreads();

  int buf = 0;
  for (int kc = 0; kc < K; kc += KCH) {
    if (kc + KCH < K) stage(kc + KCH, buf ^ 1);  // overlap with compute

    const unsigned short* sA = &smA[buf][0];
    const unsigned short* sB = &smB[buf][0];
#pragma unroll
    for (int kk = 0; kk < KCH; kk += 32) {
      v16bf a0 = load_a_frag(sA + (wr + mrow) * APITCH, kk, hi);
      v16bf a1 = load_a_frag(sA + (wr + 16 + mrow) * APITCH, kk, hi);
      v16bf b0 = load_bt_frag(sB + (wc + mrow) * APITCH, kk, hi);
      v16bf b1 = load_bt_frag(sB + (wc + 16 + mrow) * APITCH, kk, hi);
      v16bf b2 = load_bt_frag(sB + (wc + 32 + mrow) * APITCH, kk, hi);
      v16bf b3 = load_bt_frag(sB + (wc + 48 + mrow) * APITCH, kk, hi);
      acc[0][0] = wmma_bf16(a0, b0, acc[0][0]);
      acc[0][1] = wmma_bf16(a0, b1, acc[0][1]);
      acc[0][2] = wmma_bf16(a0, b2, acc[0][2]);
      acc[0][3] = wmma_bf16(a0, b3, acc[0][3]);
      acc[1][0] = wmma_bf16(a1, b0, acc[1][0]);
      acc[1][1] = wmma_bf16(a1, b1, acc[1][1]);
      acc[1][2] = wmma_bf16(a1, b2, acc[1][2]);
      acc[1][3] = wmma_bf16(a1, b3, acc[1][3]);
    }

    wait_async0();     // next buffer staged (and all waves past this chunk)
    __syncthreads();
    buf ^= 1;
  }

  float* geb = GE + (size_t)b * NN * NN;
#pragma unroll
  for (int i = 0; i < 2; ++i)
#pragma unroll
    for (int j = 0; j < 4; ++j) {
      const int row0 = R0 + wr + i * 16 + hi * 8;
      const int col  = C0 + wc + j * 16 + mrow;
#pragma unroll
      for (int r = 0; r < 8; ++r)
        geb[(size_t)(row0 + r) * NN + col] = acc[i][j][r];
    }
}

// ---------------------------------------------------------------------------
// Kernel 4: standalone row softmax: f32 (N wide) -> bf16, row per block.
// ---------------------------------------------------------------------------
__global__ __launch_bounds__(256) void gpam_softmax(
    const float* __restrict__ GE, unsigned short* __restrict__ out) {
  const int b   = blockIdx.y;
  const int row = blockIdx.x;
  const int tid = threadIdx.x;
  const float* gr = GE + ((size_t)b * NN + row) * NN;

  __shared__ float red[256];

  float v[16];
  float m = -3.402823466e38f;
#pragma unroll
  for (int i = 0; i < 16; ++i) {
    v[i] = gr[tid + i * 256];
    m = fmaxf(m, v[i]);
  }
  red[tid] = m;
  __syncthreads();
#pragma unroll
  for (int s = 128; s > 0; s >>= 1) {
    if (tid < s) red[tid] = fmaxf(red[tid], red[tid + s]);
    __syncthreads();
  }
  m = red[0];
  __syncthreads();

  float s = 0.0f;
#pragma unroll
  for (int i = 0; i < 16; ++i) {
    v[i] = __expf(v[i] - m);
    s += v[i];
  }
  red[tid] = s;
  __syncthreads();
#pragma unroll
  for (int st = 128; st > 0; st >>= 1) {
    if (tid < st) red[tid] += red[tid + st];
    __syncthreads();
  }
  const float inv = 1.0f / red[0];

  unsigned short* orow = out + ((size_t)b * NN + row) * NN;
#pragma unroll
  for (int i = 0; i < 16; ++i) orow[tid + i * 256] = f2bf(v[i] * inv);
}

// ---------------------------------------------------------------------------
// Kernel 5: outT = GA x V^T (NT), epilogue out = gamma*outT^T + x (f32).
// ---------------------------------------------------------------------------
__global__ __launch_bounds__(128) void gpam_out(
    const unsigned short* __restrict__ GA, const unsigned short* __restrict__ vcn,
    const float* __restrict__ x, const float* __restrict__ gamma,
    float* __restrict__ out) {
  const int b  = blockIdx.y;
  const int r0 = blockIdx.x * 16;
  const int tid  = threadIdx.x;
  const int lane = tid & 31;
  const int wave = tid >> 5;
  const int mrow = lane & 15;
  const int hi   = lane >> 4;

  const unsigned short* pa =
      GA + (size_t)b * NN * NN + (size_t)(r0 + mrow) * NN;
  const int c0 = wave * 16;
  const unsigned short* pb =
      vcn + (size_t)b * CC * NN + (size_t)(c0 + mrow) * NN;

  v8f acc;
#pragma unroll
  for (int r = 0; r < 8; ++r) acc[r] = 0.0f;

  for (int kk = 0; kk < NN; kk += 32) {
    v16bf af = load_a_frag(pa, kk, hi);
    v16bf bf = load_bt_frag(pb, kk, hi);
    acc = wmma_bf16(af, bf, acc);
  }

  const float gm = gamma[0];
  const int c = c0 + mrow;
#pragma unroll
  for (int r = 0; r < 8; ++r) {
    const int n = r0 + r + hi * 8;
    const size_t o = ((size_t)b * CC + c) * (size_t)NN + n;
    out[o] = fmaf(gm, acc[r], x[o]);
  }
}

// ---------------------------------------------------------------------------
// Launch
// ---------------------------------------------------------------------------
extern "C" void kernel_launch(void* const* d_in, const int* in_sizes, int n_in,
                              void* d_out, int out_size, void* d_ws, size_t ws_size,
                              hipStream_t stream) {
  (void)in_sizes; (void)n_in; (void)out_size; (void)ws_size;

  const float* x    = (const float*)d_in[0];
  const float* g    = (const float*)d_in[1];
  const float* Wq   = (const float*)d_in[2];
  const float* bq   = (const float*)d_in[3];
  const float* Wk   = (const float*)d_in[4];
  const float* bk   = (const float*)d_in[5];
  const float* Wv   = (const float*)d_in[6];
  const float* bv   = (const float*)d_in[7];
  const float* Wqg  = (const float*)d_in[8];
  const float* bqg  = (const float*)d_in[9];
  const float* Wkg  = (const float*)d_in[10];
  const float* bkg  = (const float*)d_in[11];
  const float* gmma = (const float*)d_in[12];
  float* out = (float*)d_out;

  unsigned char* ws = (unsigned char*)d_ws;
  size_t off = 0;
  auto carve = [&](size_t bytes) {
    unsigned char* p = ws + off;
    off += (bytes + 255) & ~(size_t)255;
    return p;
  };
  unsigned short* qt    = (unsigned short*)carve((size_t)NB * NN * CC  * 2);
  unsigned short* kt    = (unsigned short*)carve((size_t)NB * NN * CC  * 2);
  unsigned short* vcn   = (unsigned short*)carve((size_t)NB * CC * NN  * 2);
  unsigned short* qgt   = (unsigned short*)carve((size_t)NB * NN * CGC * 2);
  unsigned short* kgt   = (unsigned short*)carve((size_t)NB * NN * CGC * 2);
  unsigned short* att   = (unsigned short*)carve((size_t)NB * NN * NN  * 2);
  unsigned short* attgT = (unsigned short*)carve((size_t)NB * NN * NN  * 2);
  unsigned short* ga    = (unsigned short*)carve((size_t)NB * NN * NN  * 2);
  float*          ge    = (float*)carve((size_t)NB * NN * NN * 4);

  // 1) projections
  gpam_proj<<<(NB * NN) / 256, 256, 0, stream>>>(
      x, g, Wq, bq, Wk, bk, Wv, bv, Wqg, bqg, Wkg, bkg,
      qt, kt, vcn, qgt, kgt);

  const size_t smem = (16 * NN + 16 * 16 + 16) * sizeof(float);  // ~257 KB

  // 2) attention    = softmax(Q^T K)       K = 64  -> row-major bf16
  gpam_gemm_softmax<<<dim3(NN / 16, NB), 256, smem, stream>>>(
      qt, kt, att, CC, 0);
  // 3) attention_g  = softmax(Qg^T Kg)     K = 32  -> stored TRANSPOSED
  gpam_gemm_softmax<<<dim3(NN / 16, NB), 256, smem, stream>>>(
      qgt, kgt, attgT, CGC, 1);
  // 4) guide energy = att x att_g          K = 4096 (dominant ~550 GFLOP)
  gpam_gemm_blocked<<<dim3(NN / 128, NN / 128, NB), 256, 0, stream>>>(
      att, attgT, ge, NN);
  //    guide attention = row softmax(ge) -> bf16
  gpam_softmax<<<dim3(NN, NB), 256, 0, stream>>>(ge, ga);
  // 5) out = gamma * (GA x V^T)^T + x
  gpam_out<<<dim3(NN / 16, NB), 128, 0, stream>>>(ga, vcn, x, gmma, out);
}
